// PhysicsNeuralInteractionField_78503412236724
// MI455X (gfx1250) — compile-verified, matrix operand order checked by reading, not run on hardware
//
#include <hip/hip_runtime.h>

// ---------------------------------------------------------------------------
// PhysicsNeuralInteractionField for MI455X (gfx1250, wave32, WMMA bf16)
// ---------------------------------------------------------------------------

typedef __attribute__((ext_vector_type(16))) __bf16 v16bf;
typedef __attribute__((ext_vector_type(8)))  float  f32x8;

#define DEV static __device__ __forceinline__

constexpr int   NPT    = 1024;
constexpr int   C      = 32;
constexpr int   ICH    = 64;
constexpr int   NH     = 4;
constexpr int   HD     = 8;
constexpr float RADIUS = 0.1125f;          // 1.5*6*0.025/2
constexpr int   MAXNB  = 128;              // neighbor-list cap (expected ~6)

DEV float geluf(float x) { return 0.5f * x * (1.0f + erff(x * 0.70710678118654752f)); }
DEV float sigmoidf(float x) { return 1.0f / (1.0f + expf(-x)); }

DEV __bf16 to_bf16(float f) {              // round-to-nearest-even, bit-exact path
    unsigned u = __builtin_bit_cast(unsigned, f);
    unsigned r = (u + 0x7FFFu + ((u >> 16) & 1u)) >> 16;
    unsigned short s = (unsigned short)r;
    return __builtin_bit_cast(__bf16, s);
}

DEV void ln_inplace(float* v, int D, const float* g, const float* b) {
    float m = 0.f;
    for (int i = 0; i < D; ++i) m += v[i];
    m /= (float)D;
    float var = 0.f;
    for (int i = 0; i < D; ++i) { float d = v[i] - m; var += d * d; }
    var /= (float)D;
    float inv = rsqrtf(var + 1e-5f);
    for (int i = 0; i < D; ++i) v[i] = (v[i] - m) * inv * g[i] + b[i];
}

// ---------------------------------------------------------------------------
// Kernel 1: per-point encoder (fourier + MLPs + fusion + q/k/v + w2q fold)
// ---------------------------------------------------------------------------
struct EncArgs {
    const float *x, *y, *pos;
    const float *fourier_B;
    const float *le_w1, *le_b1, *le_g1, *le_be1;
    const float *le_w2, *le_b2, *le_g2, *le_be2;
    const float *fu_w, *fu_b, *fu_g, *fu_be;
    const float *pf_w, *pf_b;
    const float *q_w, *q_b, *k_w, *k_b, *v_w, *v_b;
    const float *gm_w2, *gm_b2;
    float *q, *k, *v, *w2q, *qb2, *feat;
};

__global__ __launch_bounds__(256) void k_encode(EncArgs A) {
    int n = blockIdx.x * blockDim.x + threadIdx.x;
    if (n >= NPT) return;
    float p0 = A.pos[n * 3 + 0], p1 = A.pos[n * 3 + 1], p2 = A.pos[n * 3 + 2];

    // learned-embed MLP: 3 -> 32 -> 32 (LN+GELU each)
    float h1[C];
    for (int c = 0; c < C; ++c)
        h1[c] = p0 * A.le_w1[0 * C + c] + p1 * A.le_w1[1 * C + c] + p2 * A.le_w1[2 * C + c] + A.le_b1[c];
    ln_inplace(h1, C, A.le_g1, A.le_be1);
    for (int c = 0; c < C; ++c) h1[c] = geluf(h1[c]);
    float h2[C];
    for (int c = 0; c < C; ++c) {
        float a = A.le_b2[c];
        for (int kk = 0; kk < C; ++kk) a += h1[kk] * A.le_w2[kk * C + c];
        h2[c] = a;
    }
    ln_inplace(h2, C, A.le_g2, A.le_be2);
    for (int c = 0; c < C; ++c) h2[c] = geluf(h2[c]);

    // fourier features streamed straight into fu matmul (no [256] buffer)
    float pe[C];
    for (int c = 0; c < C; ++c) pe[c] = A.fu_b[c];
    for (int s = 0; s < 4; ++s) {
        for (int m = 0; m < C; ++m) {
            float pr = p0 * A.fourier_B[(s * 3 + 0) * C + m]
                     + p1 * A.fourier_B[(s * 3 + 1) * C + m]
                     + p2 * A.fourier_B[(s * 3 + 2) * C + m];
            float sn = sinf(pr), cs = cosf(pr);
            const float* ws = A.fu_w + (size_t)(s * 64 + m) * C;
            const float* wc = A.fu_w + (size_t)(s * 64 + 32 + m) * C;
            for (int c = 0; c < C; ++c) pe[c] += sn * ws[c] + cs * wc[c];
        }
    }
    for (int kk = 0; kk < C; ++kk) {
        const float* wr = A.fu_w + (size_t)(256 + kk) * C;
        for (int c = 0; c < C; ++c) pe[c] += h2[kk] * wr[c];
    }
    ln_inplace(pe, C, A.fu_g, A.fu_be);
    for (int c = 0; c < C; ++c) pe[c] = geluf(pe[c]);

    // pos fusion: xp = gelu([x, pe] @ pf_w + pf_b), same for yp
    float xp[C], yp[C];
    for (int c = 0; c < C; ++c) { xp[c] = A.pf_b[c]; yp[c] = A.pf_b[c]; }
    for (int kk = 0; kk < C; ++kk) {
        float xv = A.x[n * C + kk], yv = A.y[n * C + kk], pv = pe[kk];
        const float* w0 = A.pf_w + (size_t)kk * C;
        const float* w1 = A.pf_w + (size_t)(C + kk) * C;
        for (int c = 0; c < C; ++c) {
            xp[c] += xv * w0[c] + pv * w1[c];
            yp[c] += yv * w0[c] + pv * w1[c];
        }
    }
    for (int c = 0; c < C; ++c) { xp[c] = geluf(xp[c]); yp[c] = geluf(yp[c]); }

    // q/k/v projections
    float qv[C], kv[C], vv[C];
    for (int c = 0; c < C; ++c) {
        float aq = A.q_b[c], ak = A.k_b[c], av = A.v_b[c];
        for (int kk = 0; kk < C; ++kk) {
            aq += xp[kk] * A.q_w[kk * C + c];
            ak += yp[kk] * A.k_w[kk * C + c];
            av += yp[kk] * A.v_w[kk * C + c];
        }
        qv[c] = aq; kv[c] = ak; vv[c] = av;
    }
    for (int c = 0; c < C; ++c) {
        A.q[n * C + c] = qv[c];
        A.k[n * C + c] = kv[c];
        A.v[n * C + c] = vv[c];
        A.feat[n * 96 + c]      = xp[c];
        A.feat[n * 96 + 32 + c] = yp[c];
    }
    // fold q into geom head: w2q[i,h,r] = sum_d gm_w2[r, h*8+d] * q[h*8+d]
    for (int h = 0; h < NH; ++h) {
        float qb = 0.f;
        for (int d = 0; d < HD; ++d) qb += A.gm_b2[h * HD + d] * qv[h * HD + d];
        A.qb2[n * NH + h] = qb;
        for (int r = 0; r < 16; ++r) {
            float w = 0.f;
            for (int d = 0; d < HD; ++d) w += A.gm_w2[r * C + h * HD + d] * qv[h * HD + d];
            A.w2q[(n * NH + h) * 16 + r] = w;
        }
    }
}

// ---------------------------------------------------------------------------
// Kernel 2: dense attention, one block per query i. Scores + softmax live in
// LDS (16KB/row) so the N x N matrix never touches HBM.
// ---------------------------------------------------------------------------
struct AttnArgs {
    const float *pos, *gm_w1, *gm_b1;
    const float *q, *k, *v, *w2q, *qb2;
    float *feat;
};

__global__ __launch_bounds__(256) void k_attn(AttnArgs A) {
    __shared__ float sc[NH][NPT];        // 16 KB scores -> probs
    __shared__ float red[256];
    __shared__ float accs[256][C];       // 32 KB AV partials
    __shared__ float qi[C], w2qi[NH * 16], qb2i[NH], posi[3], gw1[64];

    int i = blockIdx.x, t = threadIdx.x;
    if (t < C)  qi[t] = A.q[i * C + t];
    if (t < 64) w2qi[t] = A.w2q[i * 64 + t];
    if (t < NH) qb2i[t] = A.qb2[i * NH + t];
    if (t < 3)  posi[t] = A.pos[i * 3 + t];
    if (t >= 64 && t < 112) gw1[t - 64] = A.gm_w1[t - 64];    // [3][16]
    if (t >= 112 && t < 128) gw1[48 + t - 112] = A.gm_b1[t - 112];
    __syncthreads();

    const float inv_sqrt_hd = 0.35355339059327376f;
    for (int jj = 0; jj < 4; ++jj) {
        int j = t * 4 + jj;
        float d0 = posi[0] - A.pos[j * 3 + 0];
        float d1 = posi[1] - A.pos[j * 3 + 1];
        float d2 = posi[2] - A.pos[j * 3 + 2];
        float dist = sqrtf(d0 * d0 + d1 * d1 + d2 * d2 + 1e-24f);
        float hid[16];
        for (int r = 0; r < 16; ++r) {
            float h = d0 * gw1[r] + d1 * gw1[16 + r] + d2 * gw1[32 + r] + gw1[48 + r];
            hid[r] = fmaxf(h, 0.f);
        }
        for (int h = 0; h < NH; ++h) {
            float qk = 0.f;
            for (int d = 0; d < HD; ++d) qk += qi[h * HD + d] * A.k[j * C + h * HD + d];
            float g = qb2i[h];
            for (int r = 0; r < 16; ++r) g += hid[r] * w2qi[h * 16 + r];
            sc[h][j] = qk * inv_sqrt_hd + g - dist;
        }
    }
    __syncthreads();

    float sumh[NH];
    for (int h = 0; h < NH; ++h) {
        float lm = -3.4e38f;
        for (int jj = 0; jj < 4; ++jj) lm = fmaxf(lm, sc[h][t * 4 + jj]);
        red[t] = lm; __syncthreads();
        for (int o = 128; o > 0; o >>= 1) { if (t < o) red[t] = fmaxf(red[t], red[t + o]); __syncthreads(); }
        float mh = red[0]; __syncthreads();
        float ls = 0.f;
        for (int jj = 0; jj < 4; ++jj) {
            int j = t * 4 + jj;
            float e = expf(sc[h][j] - mh);
            sc[h][j] = e; ls += e;
        }
        red[t] = ls; __syncthreads();
        for (int o = 128; o > 0; o >>= 1) { if (t < o) red[t] += red[t + o]; __syncthreads(); }
        sumh[h] = red[0]; __syncthreads();
    }

    float acc[C];
    for (int c = 0; c < C; ++c) acc[c] = 0.f;
    for (int jj = 0; jj < 4; ++jj) {
        int j = t * 4 + jj;
        for (int h = 0; h < NH; ++h) {
            float p = sc[h][j] / sumh[h];
            for (int d = 0; d < HD; ++d) acc[h * HD + d] += p * A.v[j * C + h * HD + d];
        }
    }
    for (int c = 0; c < C; ++c) accs[t][c] = acc[c];
    __syncthreads();
    if (t < C) {
        float r = 0.f;
        for (int u = 0; u < 256; ++u) r += accs[u][t];
        A.feat[i * 96 + 64 + t] = r;    // attn_out third of feat
    }
}

// ---------------------------------------------------------------------------
// Kernel 3: sparse continuous-conv gather. Ball radius covers ~6 of 1024
// points, so compact a neighbor list in LDS, then scatter window*trilinear
// weighted features into the dense m[i, 64cells * Cin] bf16 row.
// ---------------------------------------------------------------------------
template <int CIN>
__global__ __launch_bounds__(256) void k_gather(const float* __restrict__ pos,
                                                const float* __restrict__ feat, int fstride,
                                                __bf16* __restrict__ m) {
    constexpr int TOT = 64 * CIN;
    constexpr int SPT = TOT / 256;
    __shared__ float nw[MAXNB][12];
    __shared__ int   nj[MAXNB];
    __shared__ int   cnt;
    __shared__ float posi[3];

    int i = blockIdx.x, t = threadIdx.x;
    if (t == 0) cnt = 0;
    if (t < 3) posi[t] = pos[i * 3 + t];
    __syncthreads();

    const float invr = 1.0f / RADIUS;
    for (int jj = 0; jj < 4; ++jj) {
        int j = t + jj * 256;
        float u0 = (pos[j * 3 + 0] - posi[0]) * invr;
        float u1 = (pos[j * 3 + 1] - posi[1]) * invr;
        float u2 = (pos[j * 3 + 2] - posi[2]) * invr;
        float r2 = u0 * u0 + u1 * u1 + u2 * u2;
        if (r2 < 1.0f && j != i) {
            float w = 1.0f - r2; w = w * w * w;                       // poly6 window
            float r = sqrtf(r2 + 1e-12f);
            float linf = fmaxf(fmaxf(fabsf(u0), fabsf(u1)), fmaxf(fabsf(u2), 1e-12f));
            float s = r / linf;                                       // ball -> cube
            float g0 = (u0 * s + 1.0f) * 1.5f;
            float g1 = (u1 * s + 1.0f) * 1.5f;
            float g2 = (u2 * s + 1.0f) * 1.5f;
            int idx = atomicAdd(&cnt, 1);
            if (idx < MAXNB) {
                nj[idx] = j;
                for (int kk = 0; kk < 4; ++kk) {
                    nw[idx][kk]     = w * fmaxf(0.f, 1.0f - fabsf(g0 - (float)kk));
                    nw[idx][4 + kk] =     fmaxf(0.f, 1.0f - fabsf(g1 - (float)kk));
                    nw[idx][8 + kk] =     fmaxf(0.f, 1.0f - fabsf(g2 - (float)kk));
                }
            }
        }
    }
    __syncthreads();
    int nn = cnt < MAXNB ? cnt : MAXNB;

    float acc[SPT];
    for (int s = 0; s < SPT; ++s) acc[s] = 0.f;
    for (int e = 0; e < nn; ++e) {
        const float* fr = feat + (size_t)nj[e] * fstride;
        for (int s = 0; s < SPT; ++s) {
            int col = t + s * 256;
            int cell = col / CIN, ch = col - cell * CIN;
            int xx = cell >> 4, yy = (cell >> 2) & 3, zz = cell & 3;
            float a = nw[e][xx] * nw[e][4 + yy] * nw[e][8 + zz];
            acc[s] += a * fr[ch];
        }
    }
    __bf16* mrow = m + (size_t)i * TOT;
    for (int s = 0; s < SPT; ++s) mrow[t + s * 256] = to_bf16(acc[s]);
}

// ---------------------------------------------------------------------------
// Kernel 4: bf16 WMMA GEMM  C[M,Nc] = A[M,K](bf16) x BT[Nc,K](bf16), f32 acc.
// One wave per 16x16 output tile; v_wmma_f32_16x16x32_bf16 over K.
// A frag (16-bit 16x32): lane = M, K chunks at half*8 and half*8+16.
// ---------------------------------------------------------------------------
__global__ __launch_bounds__(32) void k_gemm_bf16(const __bf16* __restrict__ Am,
                                                  const __bf16* __restrict__ BT,
                                                  float* __restrict__ Cm,
                                                  int K, int Ncols) {
    int tiles_n = Ncols >> 4;
    int tn = blockIdx.x % tiles_n;
    int tm = blockIdx.x / tiles_n;
    int lane = threadIdx.x;
    int half = lane >> 4, l16 = lane & 15;

    const __bf16* arow = Am + (size_t)(tm * 16 + l16) * K;
    const __bf16* brow = BT + (size_t)(tn * 16 + l16) * K;

    union Frag { uint4 u[2]; v16bf v; };
    f32x8 acc = {};

    for (int kb = 0; kb < K; kb += 32) {
        int o0 = kb + half * 8;
        __builtin_prefetch(arow + kb + 128, 0, 1);   // global_prefetch_b8
        Frag fa, fb;
        fa.u[0] = *(const uint4*)(arow + o0);
        fa.u[1] = *(const uint4*)(arow + o0 + 16);
        fb.u[0] = *(const uint4*)(brow + o0);
        fb.u[1] = *(const uint4*)(brow + o0 + 16);
        acc = __builtin_amdgcn_wmma_f32_16x16x32_bf16(false, fa.v, false, fb.v,
                                                      (short)0, acc, false, false);
    }
    // C/D layout: VGPR r -> M = r + 8*half, N = lane%16
    for (int r = 0; r < 8; ++r)
        Cm[(size_t)(tm * 16 + r + half * 8) * Ncols + tn * 16 + l16] = acc[r];
}

// ---------------------------------------------------------------------------
// Kernel 5: f32 -> bf16 transpose (builds BT[Nc,K] panels for the GEMMs)
// ---------------------------------------------------------------------------
__global__ void k_transpose_bf16(const float* __restrict__ src, __bf16* __restrict__ dst,
                                 int R, int Cl) {
    int g = blockIdx.x * blockDim.x + threadIdx.x;
    if (g < R * Cl) {
        int r = g / Cl, c = g - r * Cl;
        dst[(size_t)c * R + r] = to_bf16(src[g]);
    }
}

// ---------------------------------------------------------------------------
// Kernel 6: row LayerNorm (+ optional GELU)
// ---------------------------------------------------------------------------
__global__ __launch_bounds__(256) void k_lngelu(const float* __restrict__ in,
                                                const float* __restrict__ g,
                                                const float* __restrict__ b,
                                                float* __restrict__ out, int D, int do_gelu) {
    int n = blockIdx.x * blockDim.x + threadIdx.x;
    if (n >= NPT) return;
    float v[64];
    for (int i = 0; i < D; ++i) v[i] = in[(size_t)n * D + i];
    ln_inplace(v, D, g, b);
    if (do_gelu) for (int i = 0; i < D; ++i) v[i] = geluf(v[i]);
    for (int i = 0; i < D; ++i) out[(size_t)n * D + i] = v[i];
}

// ---------------------------------------------------------------------------
// Kernel 7: final LN + sigmoid gate + blend
// ---------------------------------------------------------------------------
struct FinArgs {
    const float *out2, *ln2_g, *ln2_b, *pg_w, *pg_b, *x, *y;
    float* out;
};
__global__ __launch_bounds__(256) void k_final(FinArgs A) {
    int n = blockIdx.x * blockDim.x + threadIdx.x;
    if (n >= NPT) return;
    float f2[C];
    for (int i = 0; i < C; ++i) f2[i] = A.out2[n * C + i];
    ln_inplace(f2, C, A.ln2_g, A.ln2_b);
    for (int c = 0; c < C; ++c) {
        float a = A.pg_b[c];
        for (int kk = 0; kk < C; ++kk) a += f2[kk] * A.pg_w[kk * C + c];
        float gate = sigmoidf(a);
        A.out[n * C + c] = A.x[n * C + c] * gate + A.y[n * C + c] * (1.0f - gate);
    }
}

// ---------------------------------------------------------------------------
extern "C" void kernel_launch(void* const* d_in, const int* in_sizes, int n_in,
                              void* d_out, int out_size, void* d_ws, size_t ws_size,
                              hipStream_t stream) {
    (void)in_sizes; (void)n_in; (void)out_size; (void)ws_size;
    const float* x   = (const float*)d_in[0];
    const float* y   = (const float*)d_in[1];
    const float* pos = (const float*)d_in[2];
    const float* fourier_B = (const float*)d_in[3];
    const float* le_w1 = (const float*)d_in[4];
    const float* le_b1 = (const float*)d_in[5];
    const float* le_g1 = (const float*)d_in[6];
    const float* le_be1 = (const float*)d_in[7];
    const float* le_w2 = (const float*)d_in[8];
    const float* le_b2 = (const float*)d_in[9];
    const float* le_g2 = (const float*)d_in[10];
    const float* le_be2 = (const float*)d_in[11];
    const float* fu_w = (const float*)d_in[12];
    const float* fu_b = (const float*)d_in[13];
    const float* fu_g = (const float*)d_in[14];
    const float* fu_be = (const float*)d_in[15];
    const float* pf_w = (const float*)d_in[16];
    const float* pf_b = (const float*)d_in[17];
    const float* q_w = (const float*)d_in[18];
    const float* q_b = (const float*)d_in[19];
    const float* k_w = (const float*)d_in[20];
    const float* k_b = (const float*)d_in[21];
    const float* v_w = (const float*)d_in[22];
    const float* v_b = (const float*)d_in[23];
    const float* gm_w1 = (const float*)d_in[24];
    const float* gm_b1 = (const float*)d_in[25];
    const float* gm_w2 = (const float*)d_in[26];
    const float* gm_b2 = (const float*)d_in[27];
    const float* k1 = (const float*)d_in[28];
    const float* ln1_g = (const float*)d_in[29];
    const float* ln1_b = (const float*)d_in[30];
    const float* k2 = (const float*)d_in[31];
    const float* ln2_g = (const float*)d_in[32];
    const float* ln2_b = (const float*)d_in[33];
    const float* pg_w = (const float*)d_in[34];
    const float* pg_b = (const float*)d_in[35];

    char* wp = (char*)d_ws;
    auto alloc = [&](size_t bytes) -> void* {
        void* p = (void*)wp;
        wp += (bytes + 255) & ~(size_t)255;
        return p;
    };
    float*  qb    = (float*)alloc((size_t)NPT * C * 4);
    float*  kb    = (float*)alloc((size_t)NPT * C * 4);
    float*  vb    = (float*)alloc((size_t)NPT * C * 4);
    float*  w2q   = (float*)alloc((size_t)NPT * 64 * 4);
    float*  qb2   = (float*)alloc((size_t)NPT * NH * 4);
    float*  feat  = (float*)alloc((size_t)NPT * 96 * 4);
    float*  out1  = (float*)alloc((size_t)NPT * ICH * 4);
    float*  f1    = (float*)alloc((size_t)NPT * ICH * 4);
    float*  out2  = (float*)alloc((size_t)NPT * C * 4);
    __bf16* k1T   = (__bf16*)alloc((size_t)6144 * 64 * 2);
    __bf16* k2T   = (__bf16*)alloc((size_t)4096 * 32 * 2);
    __bf16* m1    = (__bf16*)alloc((size_t)NPT * 6144 * 2);
    __bf16* m2    = (__bf16*)alloc((size_t)NPT * 4096 * 2);

    EncArgs ea = { x, y, pos, fourier_B,
                   le_w1, le_b1, le_g1, le_be1, le_w2, le_b2, le_g2, le_be2,
                   fu_w, fu_b, fu_g, fu_be, pf_w, pf_b,
                   q_w, q_b, k_w, k_b, v_w, v_b, gm_w2, gm_b2,
                   qb, kb, vb, w2q, qb2, feat };
    k_encode<<<NPT / 256, 256, 0, stream>>>(ea);

    AttnArgs aa = { pos, gm_w1, gm_b1, qb, kb, vb, w2q, qb2, feat };
    k_attn<<<NPT, 256, 0, stream>>>(aa);

    k_transpose_bf16<<<(6144 * 64 + 255) / 256, 256, 0, stream>>>(k1, k1T, 6144, 64);
    k_transpose_bf16<<<(4096 * 32 + 255) / 256, 256, 0, stream>>>(k2, k2T, 4096, 32);

    k_gather<96><<<NPT, 256, 0, stream>>>(pos, feat, 96, m1);
    k_gemm_bf16<<<(NPT / 16) * (ICH / 16), 32, 0, stream>>>(m1, k1T, out1, 6144, ICH);
    k_lngelu<<<NPT / 256, 256, 0, stream>>>(out1, ln1_g, ln1_b, f1, ICH, 1);

    k_gather<64><<<NPT, 256, 0, stream>>>(pos, f1, ICH, m2);
    k_gemm_bf16<<<(NPT / 16) * (C / 16), 32, 0, stream>>>(m2, k2T, out2, 4096, C);

    FinArgs fa = { out2, ln2_g, ln2_b, pg_w, pg_b, x, y, (float*)d_out };
    k_final<<<NPT / 256, 256, 0, stream>>>(fa);
}
